// BaichuanAttention_58798102282689
// MI455X (gfx1250) — matrix operand order repaired and checked by reading, not compiled, not run on hardware
//
#include <hip/hip_runtime.h>
#include <hip/hip_bf16.h>

typedef __bf16 bf16_t;
typedef bf16_t v16bf __attribute__((ext_vector_type(16)));
typedef bf16_t v8bf  __attribute__((ext_vector_type(8)));
typedef bf16_t v4bf  __attribute__((ext_vector_type(4)));
typedef float  v8f   __attribute__((ext_vector_type(8)));

union Frag { v16bf v; v8bf h[2]; };

#define SEQL   2048
#define HIDDEN 5120
#define NQKV   15360
#define NH     40
#define HD     128

#define BM 128
#define BN 128
#define BK 64
#define LDT 72          // padded LDS row stride (bf16 elems) for 64-wide tiles
#define KS_STRIDE 136   // padded stride for 128-wide K tile
#define VS_STRIDE 72
#define PS_STRIDE 72

#define WMMA_BF16(a, b, c) \
  __builtin_amdgcn_wmma_f32_16x16x32_bf16(false, (a), false, (b), (short)0, (c), false, false)

// async global->LDS copy, 16B per lane; tracked by ASYNCcnt
#define ASYNC_LDS_B128(ldsaddr, gaddr)                                  \
  asm volatile("global_load_async_to_lds_b128 %0, %1, off"              \
               :: "v"(ldsaddr), "v"(gaddr) : "memory")
#define WAIT_ASYNC0 asm volatile("s_wait_asynccnt 0" ::: "memory")

// ---------------------------------------------------------------------------
// Shared GEMM mainloop, double-buffered LDS. C[128x128] tile.
// Wave grid 4(M) x 2(N): each wave computes 32x64 = 2x4 WMMA tiles.
// fp32 operands: staged through registers (reg double buffer) + convert.
// bf16 A operand (oproj): staged by async global->LDS copies (ASYNCcnt).
// ---------------------------------------------------------------------------
template <bool A_BF16>
__device__ __forceinline__ void gemm_mainloop(
    const float* __restrict__ Af, const bf16_t* __restrict__ Ab,
    const float* __restrict__ Bf, int K, int aRow0, int bRow0,
    bf16_t* a_s, bf16_t* b_s, v8f acc[2][4]) {
  const int tid  = threadIdx.x;
  const int lane = tid & 31;
  const int w    = tid >> 5;
  const int wm   = w & 3;
  const int wn   = w >> 2;
  const int m16  = lane & 15;
  const int half = lane >> 4;
  const int kq   = half * 8;    // A-fragment K sub-offset
  const int kh   = half * 8;    // B-fragment K sub-offset
  const int ldr  = tid >> 2;          // 0..63 (staging row)
  const int ldc  = (tid & 3) * 16;    // 0,16,32,48 (staging col)

  float4 ra[8], rb[8];

  auto gload = [&](int k0) {
#pragma unroll
    for (int rr = 0; rr < 2; ++rr) {
      const int row = rr * 64 + ldr;
      const float* bsrc = Bf + (size_t)(bRow0 + row) * K + k0 + ldc;
#pragma unroll
      for (int j = 0; j < 4; ++j) rb[rr * 4 + j] = *(const float4*)(bsrc + j * 4);
      if constexpr (!A_BF16) {
        const float* asrc = Af + (size_t)(aRow0 + row) * K + k0 + ldc;
#pragma unroll
        for (int j = 0; j < 4; ++j) ra[rr * 4 + j] = *(const float4*)(asrc + j * 4);
      }
    }
  };

  auto issue_asyncA = [&](int k0, int buf) {
    if constexpr (A_BF16) {
#pragma unroll
      for (int rr = 0; rr < 2; ++rr) {
        const int row = rr * 64 + ldr;
#pragma unroll
        for (int h = 0; h < 2; ++h) {
          unsigned laddr =
              (unsigned)(size_t)&a_s[buf * (BM * LDT) + row * LDT + ldc + h * 8];
          const bf16_t* g = Ab + (size_t)(aRow0 + row) * K + k0 + ldc + h * 8;
          ASYNC_LDS_B128(laddr, (unsigned long long)(size_t)g);
        }
      }
    }
  };

  auto sstore = [&](int buf) {
    bf16_t* bs = b_s + buf * (BM * LDT);
#pragma unroll
    for (int rr = 0; rr < 2; ++rr) {
      const int row = rr * 64 + ldr;
#pragma unroll
      for (int j = 0; j < 4; ++j) {
        float4 f = rb[rr * 4 + j];
        v4bf bv;
        bv[0] = (bf16_t)f.x; bv[1] = (bf16_t)f.y;
        bv[2] = (bf16_t)f.z; bv[3] = (bf16_t)f.w;
        *(v4bf*)&bs[row * LDT + ldc + j * 4] = bv;
      }
      if constexpr (!A_BF16) {
        bf16_t* as = a_s + buf * (BM * LDT);
#pragma unroll
        for (int j = 0; j < 4; ++j) {
          float4 f = ra[rr * 4 + j];
          v4bf bv;
          bv[0] = (bf16_t)f.x; bv[1] = (bf16_t)f.y;
          bv[2] = (bf16_t)f.z; bv[3] = (bf16_t)f.w;
          *(v4bf*)&as[row * LDT + ldc + j * 4] = bv;
        }
      }
    }
  };

  auto compute = [&](int buf) {
    const bf16_t* as = a_s + buf * (BM * LDT);
    const bf16_t* bs = b_s + buf * (BM * LDT);
#pragma unroll
    for (int kk = 0; kk < 2; ++kk) {
      Frag a0, a1, bt[4];
      const int abase0 = (wm * 32 + m16) * LDT + kk * 32 + kq;
      a0.h[0] = *(const v8bf*)&as[abase0];
      a0.h[1] = *(const v8bf*)&as[abase0 + 16];
      const int abase1 = abase0 + 16 * LDT;
      a1.h[0] = *(const v8bf*)&as[abase1];
      a1.h[1] = *(const v8bf*)&as[abase1 + 16];
#pragma unroll
      for (int t = 0; t < 4; ++t) {
        const int bbase = (wn * 64 + t * 16 + m16) * LDT + kk * 32 + kh;
        bt[t].h[0] = *(const v8bf*)&bs[bbase];
        bt[t].h[1] = *(const v8bf*)&bs[bbase + 16];
      }
#pragma unroll
      for (int t = 0; t < 4; ++t) {
        acc[0][t] = WMMA_BF16(a0.v, bt[t].v, acc[0][t]);
        acc[1][t] = WMMA_BF16(a1.v, bt[t].v, acc[1][t]);
      }
    }
  };

  const int niter = K / BK;
  issue_asyncA(0, 0);
  gload(0);
  for (int i = 0; i < niter; ++i) {
    sstore(i & 1);                // regs for tile i -> LDS buf i&1 (safe: last
                                  // read of this buf was compute(i-2))
    if (i + 1 < niter) gload((i + 1) * BK);          // overlap with compute(i)
    if (i + 2 < niter)                               // warm L2 for tile i+2
      __builtin_prefetch(Bf + (size_t)(bRow0 + ldr) * K + (i + 2) * BK + ldc, 0, 1);
    if constexpr (A_BF16) { WAIT_ASYNC0; }           // asyncA(i) landed (this wave)
    __syncthreads();                                 // all waves: tile i ready
    if constexpr (A_BF16) {
      if (i + 1 < niter) issue_asyncA((i + 1) * BK, (i + 1) & 1);  // overlap
    }
    compute(i & 1);
  }
}

// ---------------------------------------------------------------------------
// Kernel 1: fused QKV projection. Writes Q,K as [head][s][d] bf16 and V
// transposed as [head][d][s] bf16 (so PV B-fragments are contiguous).
// ---------------------------------------------------------------------------
__global__ __launch_bounds__(256) void qkv_gemm_kernel(
    const float* __restrict__ X, const float* __restrict__ Wp,
    bf16_t* __restrict__ qws, bf16_t* __restrict__ kws, bf16_t* __restrict__ vws) {
  __shared__ bf16_t a_s[2 * BM * LDT];
  __shared__ bf16_t b_s[2 * BN * LDT];
  v8f acc[2][4];
  v8f zero = {};
  for (int i = 0; i < 2; ++i)
    for (int t = 0; t < 4; ++t) acc[i][t] = zero;

  const int bm = blockIdx.y, bn = blockIdx.x;
  gemm_mainloop<false>(X, nullptr, Wp, HIDDEN, bm * BM, bn * BN, a_s, b_s, acc);

  const int tid  = threadIdx.x;
  const int lane = tid & 31;
  const int w    = tid >> 5;
  const int wm   = w & 3, wn = w >> 2;
  const int m16  = lane & 15, half = lane >> 4;
  const int rowb = bm * BM + wm * 32;

  for (int i = 0; i < 2; ++i) {
    for (int t = 0; t < 4; ++t) {
      const int col  = bn * BN + wn * 64 + t * 16 + m16;
      const int sec  = col / HIDDEN;          // 0=Q 1=K 2=V (uniform per tile)
      const int oc   = col - sec * HIDDEN;
      const int head = oc >> 7;
      const int d    = oc & 127;
      for (int g = 0; g < 8; ++g) {
        const int row = rowb + i * 16 + g + 8 * half;
        const bf16_t bv = (bf16_t)acc[i][t][g];
        if (sec == 0)      qws[((size_t)head * SEQL + row) * HD + d] = bv;
        else if (sec == 1) kws[((size_t)head * SEQL + row) * HD + d] = bv;
        else               vws[((size_t)head * HD + d) * SEQL + row] = bv;
      }
    }
  }
}

// ---------------------------------------------------------------------------
// Kernel 2: flash attention with ALiBi + causal mask. One block = one head x
// 128 q rows; each wave owns 16 q rows. K/V double-buffered in LDS via
// async global->LDS copies, overlapped with WMMA compute.
// ---------------------------------------------------------------------------
__global__ __launch_bounds__(256) void attn_kernel(
    const bf16_t* __restrict__ qws, const bf16_t* __restrict__ kws,
    const bf16_t* __restrict__ vws, bf16_t* __restrict__ aws) {
  __shared__ bf16_t ks[2][64 * KS_STRIDE];
  __shared__ bf16_t vs[2][HD * VS_STRIDE];
  __shared__ bf16_t ps[8][16 * PS_STRIDE];

  const int qb   = blockIdx.x;
  const int head = blockIdx.y;
  const int tid  = threadIdx.x;
  const int lane = tid & 31;
  const int w    = tid >> 5;
  const int m16  = lane & 15;
  const int half = lane >> 4;
  const int kq   = half * 8;
  const int q0   = qb * 128 + w * 16;

  const float slope = (head < 32) ? exp2f(-0.25f * (float)(head + 1))
                                  : exp2f(-0.125f * (float)(2 * (head - 32) + 1));
  const float scale = 0.08838834764831845f;  // 1/sqrt(128)

  auto issue_kv = [&](int kb, int buf) {
    const int kbase = kb * 64;
    const int r  = tid >> 2;
    const int c0 = (tid & 3) * 8;
    const bf16_t* ksrc = kws + ((size_t)head * SEQL + kbase + r) * HD;
#pragma unroll
    for (int cc = 0; cc < 4; ++cc) {
      const int c = c0 + cc * 32;
      unsigned laddr = (unsigned)(size_t)&ks[buf][r * KS_STRIDE + c];
      ASYNC_LDS_B128(laddr, (unsigned long long)(size_t)(ksrc + c));
    }
    const int d  = tid >> 1;
    const int h0 = (tid & 1) * 32;
    const bf16_t* vsrc = vws + ((size_t)head * HD + d) * SEQL + kbase + h0;
#pragma unroll
    for (int cc = 0; cc < 4; ++cc) {
      unsigned laddr = (unsigned)(size_t)&vs[buf][d * VS_STRIDE + h0 + cc * 8];
      ASYNC_LDS_B128(laddr, (unsigned long long)(size_t)(vsrc + cc * 8));
    }
  };

  // Q fragments for this wave's 16 rows (16x128), resident in registers.
  Frag qf[4];
  const bf16_t* qrow = qws + ((size_t)head * SEQL + q0 + m16) * HD;
#pragma unroll
  for (int kk = 0; kk < 4; ++kk) {
    qf[kk].h[0] = *(const v8bf*)(qrow + kk * 32 + kq);
    qf[kk].h[1] = *(const v8bf*)(qrow + kk * 32 + kq + 16);
  }

  float mprev[8], lsum[8];
  v8f o[8];
  v8f zero = {};
  for (int g = 0; g < 8; ++g) { mprev[g] = -3e38f; lsum[g] = 0.f; }
  for (int j = 0; j < 8; ++j) o[j] = zero;

  const int nkb = (qb + 1) * 2;  // 64-key blocks up to the causal frontier
  issue_kv(0, 0);
  for (int kb = 0; kb < nkb; ++kb) {
    const int kbase = kb * 64;
    const int buf   = kb & 1;
    WAIT_ASYNC0;          // this wave's loads for block kb have landed
    __syncthreads();      // all waves: block kb ready; all done with kb-1
    if (kb + 1 < nkb) issue_kv(kb + 1, buf ^ 1);  // overlap with compute(kb)

    // ---- S = Q K^T (16 x 64), fp32 accum ----
    v8f s[4];
    for (int t = 0; t < 4; ++t) s[t] = zero;
#pragma unroll
    for (int t = 0; t < 4; ++t) {
#pragma unroll
      for (int kk = 0; kk < 4; ++kk) {
        Frag b;
        const int addr = (t * 16 + m16) * KS_STRIDE + kk * 32 + half * 8;
        b.h[0] = *(const v8bf*)&ks[buf][addr];
        b.h[1] = *(const v8bf*)&ks[buf][addr + 16];
        s[t] = WMMA_BF16(qf[kk].v, b.v, s[t]);
      }
    }

    // ---- ALiBi + causal + online softmax ----
    float rmax[8];
    for (int g = 0; g < 8; ++g) rmax[g] = -3e38f;
#pragma unroll
    for (int t = 0; t < 4; ++t) {
      const int col = kbase + t * 16 + m16;
#pragma unroll
      for (int g = 0; g < 8; ++g) {
        const int row = q0 + g + 8 * half;
        const float val = (col <= row)
            ? (s[t][g] * scale - slope * (float)(row - col)) : -3e38f;
        s[t][g] = val;
        rmax[g] = fmaxf(rmax[g], val);
      }
    }
    for (int g = 0; g < 8; ++g)
      for (int dxor = 1; dxor < 16; dxor <<= 1)
        rmax[g] = fmaxf(rmax[g], __shfl_xor(rmax[g], dxor, 32));

    float mnew[8], corr[8], rsum[8];
    for (int g = 0; g < 8; ++g) {
      mnew[g] = fmaxf(mprev[g], rmax[g]);
      corr[g] = __expf(mprev[g] - mnew[g]);
      mprev[g] = mnew[g];
      rsum[g] = 0.f;
    }
#pragma unroll
    for (int t = 0; t < 4; ++t) {
#pragma unroll
      for (int g = 0; g < 8; ++g) {
        const float p = (s[t][g] > -1e37f) ? __expf(s[t][g] - mnew[g]) : 0.f;
        rsum[g] += p;
        ps[w][(g + 8 * half) * PS_STRIDE + t * 16 + m16] = (bf16_t)p;
      }
    }
    for (int g = 0; g < 8; ++g)
      for (int dxor = 1; dxor < 16; dxor <<= 1)
        rsum[g] += __shfl_xor(rsum[g], dxor, 32);
    for (int g = 0; g < 8; ++g) lsum[g] = lsum[g] * corr[g] + rsum[g];
    for (int j = 0; j < 8; ++j)
      for (int g = 0; g < 8; ++g) o[j][g] *= corr[g];

    // wave-local P writes must land before A-fragment reads
    asm volatile("s_wait_dscnt 0" ::: "memory");

    // ---- O += P V (16 x 128) ----
#pragma unroll
    for (int kk2 = 0; kk2 < 2; ++kk2) {
      Frag ap;
      const int paddr = m16 * PS_STRIDE + kk2 * 32 + kq;
      ap.h[0] = *(const v8bf*)&ps[w][paddr];
      ap.h[1] = *(const v8bf*)&ps[w][paddr + 16];
#pragma unroll
      for (int j = 0; j < 8; ++j) {
        Frag bv;
        const int vaddr = (j * 16 + m16) * VS_STRIDE + kk2 * 32 + half * 8;
        bv.h[0] = *(const v8bf*)&vs[buf][vaddr];
        bv.h[1] = *(const v8bf*)&vs[buf][vaddr + 16];
        o[j] = WMMA_BF16(ap.v, bv.v, o[j]);
      }
    }
  }

  // ---- normalize and write attention output [s][head*128+d] as bf16 ----
  for (int j = 0; j < 8; ++j) {
    const int col = head * HD + j * 16 + m16;
    for (int g = 0; g < 8; ++g) {
      const int row = q0 + g + 8 * half;
      const float ov = o[j][g] / fmaxf(lsum[g], 1e-30f);
      aws[(size_t)row * HIDDEN + col] = (bf16_t)ov;
    }
  }
}

// ---------------------------------------------------------------------------
// Kernel 3: output projection, fp32 result. A (bf16) staged via async copies.
// ---------------------------------------------------------------------------
__global__ __launch_bounds__(256) void oproj_gemm_kernel(
    const bf16_t* __restrict__ Aattn, const float* __restrict__ Wo,
    float* __restrict__ out) {
  __shared__ bf16_t a_s[2 * BM * LDT];
  __shared__ bf16_t b_s[2 * BN * LDT];
  v8f acc[2][4];
  v8f zero = {};
  for (int i = 0; i < 2; ++i)
    for (int t = 0; t < 4; ++t) acc[i][t] = zero;

  const int bm = blockIdx.y, bn = blockIdx.x;
  gemm_mainloop<true>(nullptr, Aattn, Wo, HIDDEN, bm * BM, bn * BN, a_s, b_s, acc);

  const int tid  = threadIdx.x;
  const int lane = tid & 31;
  const int w    = tid >> 5;
  const int wm   = w & 3, wn = w >> 2;
  const int m16  = lane & 15, half = lane >> 4;

  for (int i = 0; i < 2; ++i) {
    for (int t = 0; t < 4; ++t) {
      const int col = bn * BN + wn * 64 + t * 16 + m16;
      for (int g = 0; g < 8; ++g) {
        const int row = bm * BM + wm * 32 + i * 16 + g + 8 * half;
        out[(size_t)row * HIDDEN + col] = acc[i][t][g];
      }
    }
  }
}

// ---------------------------------------------------------------------------
extern "C" void kernel_launch(void* const* d_in, const int* in_sizes, int n_in,
                              void* d_out, int out_size, void* d_ws, size_t ws_size,
                              hipStream_t stream) {
  const float* X  = (const float*)d_in[0];  // [1, 2048, 5120]
  const float* Wp = (const float*)d_in[1];  // [15360, 5120]
  const float* Wo = (const float*)d_in[2];  // [5120, 5120]
  float* out = (float*)d_out;               // [1, 2048, 5120]

  char* ws = (char*)d_ws;
  const size_t seg = (size_t)NH * SEQL * HD * sizeof(bf16_t);  // 20 MiB
  bf16_t* qws = (bf16_t*)(ws);
  bf16_t* kws = (bf16_t*)(ws + seg);
  bf16_t* vws = (bf16_t*)(ws + 2 * seg);
  bf16_t* aws = (bf16_t*)(ws + 3 * seg);

  dim3 blk(256);
  qkv_gemm_kernel<<<dim3(NQKV / BN, SEQL / BM), blk, 0, stream>>>(X, Wp, qws, kws, vws);
  attn_kernel<<<dim3(SEQL / 128, NH), blk, 0, stream>>>(qws, kws, vws, aws);
  oproj_gemm_kernel<<<dim3(HIDDEN / BN, SEQL / BM), blk, 0, stream>>>(aws, Wo, out);
}
